// PointerNet_53609781788996
// MI455X (gfx1250) — matrix operand order repaired and checked by reading, not compiled, not run on hardware
//
#include <hip/hip_runtime.h>
#include <math.h>

#define B_  8
#define S_  128
#define W_  128
#define SE_ 512
#define EE_ 256
#define SH_ 256
#define EH_ 128
#define L_  64
#define G_  256

typedef __attribute__((ext_vector_type(16))) _Float16 v16h;
typedef __attribute__((ext_vector_type(8)))  _Float16 v8h;
typedef __attribute__((ext_vector_type(8)))  float    v8f;

__device__ __forceinline__ float sigm_(float x) { return 1.0f / (1.0f + expf(-x)); }

// ---------------------------------------------------------------------------
// Weight pre-conversion: W (KxN f32 row-major) -> Wf f16 in WMMA B-fragment-
// major layout: Wf[((nt*(K/32)+kt)*32 + lane)*16 + e], lane=(n&15)+16*((k>>4)&1),
// e=k&15 (contiguous k order, matching the A-fragment reads). Batched via sW/sWf (elements).
// ---------------------------------------------------------------------------
__global__ void conv_wfrag(const float* __restrict__ W, _Float16* __restrict__ Wf,
                           int K, int N, long sW, long sWf)
{
    long i = (long)blockIdx.x * 256 + threadIdx.x;
    if (i >= (long)K * N) return;
    W  += (long)blockIdx.z * sW;
    Wf += (long)blockIdx.z * sWf;
    int k = (int)(i / N), n = (int)(i % N);
    int nt = n >> 4, kt = k >> 5;
    int lane = (n & 15) + (((k >> 4) & 1) << 4);
    int e = k & 15;
    Wf[((long)(nt * (K >> 5) + kt) * 32 + lane) * 16 + e] = (_Float16)W[(long)k * N + n];
}

// ---------------------------------------------------------------------------
// WMMA GEMM: C[b] = A[b](MxK f16 row-major, lda in halves) * Bf[b](f16 frag-
// major) + optional bias[N].  M%64==0, N%64==0, K%32==0, lda%8==0.
// 128 threads = 4 waves; block tile 64x64; wave tile 32x32 = 4 wmma / K-step.
// A tile (64x32 halves) moved global->LDS by global_load_async_to_lds_b128
// (ASYNCcnt), fragments read as 2x 16B ds loads; B fragments read straight
// from global as 32B v16h.
// ---------------------------------------------------------------------------
__global__ __launch_bounds__(128)
void gemm_wmma(const _Float16* __restrict__ A, const _Float16* __restrict__ Bf,
               const float* __restrict__ bias, float* __restrict__ C,
               int M, int N, int K, int lda,
               long sA, long sB, long sC)
{
    alignas(16) __shared__ _Float16 As[64][40];   // row padded to 80B (16B-aligned)
    const int tid  = threadIdx.x;
    const int lane = tid & 31;
    const int wave = tid >> 5;
    const int half = lane >> 4;
    const int row  = lane & 15;
    const int wm = wave >> 1, wn = wave & 1;
    const int tileM = blockIdx.y * 64;
    const int tileN = blockIdx.x * 64;
    A  += (long)blockIdx.z * sA;
    Bf += (long)blockIdx.z * sB;
    C  += (long)blockIdx.z * sC;
    const int Kt = K >> 5;

    const int nt0 = (tileN >> 4) + wn * 2;
    const _Float16* bp0 = Bf + (long)(nt0 + 0) * Kt * 512 + (long)lane * 16;
    const _Float16* bp1 = Bf + (long)(nt0 + 1) * Kt * 512 + (long)lane * 16;
    const int m0 = wm * 32 + row;

    // async staging: 64x32-half tile = 256 16B segments, 2 per thread
    const int seg0 = tid * 2;
    const int r0 = seg0 >> 2,       c0 = (seg0 & 3) * 8;
    const int r1 = (seg0 + 1) >> 2, c1 = ((seg0 + 1) & 3) * 8;
    const unsigned l0 = (unsigned)(size_t)&As[r0][c0];
    const unsigned l1 = (unsigned)(size_t)&As[r1][c1];
    const _Float16* ga0 = A + (long)(tileM + r0) * lda + c0;
    const _Float16* ga1 = A + (long)(tileM + r1) * lda + c1;

    v8f acc00 = {}, acc01 = {}, acc10 = {}, acc11 = {};
    for (int kt = 0; kt < Kt; ++kt) {
        unsigned long long a0 = (unsigned long long)(size_t)(ga0 + (kt << 5));
        unsigned long long a1 = (unsigned long long)(size_t)(ga1 + (kt << 5));
        asm volatile("global_load_async_to_lds_b128 %0, %1, off"
                     :: "v"(l0), "v"(a0) : "memory");
        asm volatile("global_load_async_to_lds_b128 %0, %1, off"
                     :: "v"(l1), "v"(a1) : "memory");
        asm volatile("s_wait_asynccnt 0x0" ::: "memory");
        __syncthreads();
        v8h a0lo = *(const v8h*)&As[m0][half * 16];
        v8h a0hi = *(const v8h*)&As[m0][half * 16 + 8];
        v8h a1lo = *(const v8h*)&As[m0 + 16][half * 16];
        v8h a1hi = *(const v8h*)&As[m0 + 16][half * 16 + 8];
        v16h af0 = __builtin_shufflevector(a0lo, a0hi, 0,1,2,3,4,5,6,7,8,9,10,11,12,13,14,15);
        v16h af1 = __builtin_shufflevector(a1lo, a1hi, 0,1,2,3,4,5,6,7,8,9,10,11,12,13,14,15);
        v16h bf0 = *(const v16h*)(bp0 + (long)kt * 512);
        v16h bf1 = *(const v16h*)(bp1 + (long)kt * 512);
        acc00 = __builtin_amdgcn_wmma_f32_16x16x32_f16(false, af0, false, bf0, (short)0, acc00, false, false);
        acc01 = __builtin_amdgcn_wmma_f32_16x16x32_f16(false, af0, false, bf1, (short)0, acc01, false, false);
        acc10 = __builtin_amdgcn_wmma_f32_16x16x32_f16(false, af1, false, bf0, (short)0, acc10, false, false);
        acc11 = __builtin_amdgcn_wmma_f32_16x16x32_f16(false, af1, false, bf1, (short)0, acc11, false, false);
        __syncthreads();
    }
    int gn0 = tileN + wn * 32 + row;
    int gn1 = gn0 + 16;
    float bv0 = bias ? bias[gn0] : 0.0f;
    float bv1 = bias ? bias[gn1] : 0.0f;
    int gmBase = tileM + wm * 32 + half * 8;
    #pragma unroll
    for (int i = 0; i < 8; ++i) {
        int gm0 = gmBase + i;
        int gm1 = gm0 + 16;
        C[(long)gm0 * N + gn0] = acc00[i] + bv0;
        C[(long)gm0 * N + gn1] = acc01[i] + bv1;
        C[(long)gm1 * N + gn0] = acc10[i] + bv0;
        C[(long)gm1 * N + gn1] = acc11[i] + bv1;
    }
}

// s2s[b,s,t] (f16, GEMM-A operand) = shared-entity adjacency, no self-loop, smask
__global__ __launch_bounds__(128)
void build_s2s(const int* __restrict__ matrix, const int* __restrict__ smask,
               _Float16* __restrict__ s2s)
{
    int b = blockIdx.y, s = blockIdx.x, t = threadIdx.x;
    __shared__ int rowS[W_];
    rowS[t] = matrix[((long)b * S_ + s) * W_ + t];
    __syncthreads();
    const int* rowT = matrix + ((long)b * S_ + t) * W_;
    int any = 0;
    for (int w = 0; w < W_; ++w)
        any |= (rowS[w] != 0) & (rowT[w] != 0);
    float v = (any && (t != s) && (smask[b * S_ + t] != 0)) ? 1.0f : 0.0f;
    s2s[((long)b * S_ + s) * S_ + t] = (_Float16)v;
}

// label tables: t1[m][c] = edge_emb[m] @ g1_w[384:448] + g1_b  (c<192)
//               t2[m][c] = edge_emb[m] @ g2_w[384:448] + g2_b  (c<320)
__global__ __launch_bounds__(320)
void build_tabs(const float* __restrict__ ee,
                const float* __restrict__ g1w, const float* __restrict__ g1b,
                const float* __restrict__ g2w, const float* __restrict__ g2b,
                float* __restrict__ t1, float* __restrict__ t2)
{
    int m = blockIdx.x, c = threadIdx.x;
    if (c < 192) {
        float a = g1b[c];
        for (int l = 0; l < L_; ++l)
            a += ee[m * L_ + l] * g1w[(SH_ + EH_ + l) * (EH_ + L_) + c];
        t1[m * (EH_ + L_) + c] = a;
    }
    float a2 = g2b[c];
    for (int l = 0; l < L_; ++l)
        a2 += ee[m * L_ + l] * g2w[(EH_ + SH_ + l) * (SH_ + L_) + c];
    t2[m * (SH_ + L_) + c] = a2;
}

// s_neigh_e[b,s,c] = sum_w mask * sigmoid(P1s[b,s,c]+P1e[b,w,c]+t1[m][c]) * val
__global__ __launch_bounds__(192)
void agg_sneigh_e(const int* __restrict__ matrix, const float* __restrict__ e_h,
                  const float* __restrict__ P1s, const float* __restrict__ P1e,
                  const float* __restrict__ t1, const float* __restrict__ ee,
                  float* __restrict__ out)
{
    int b = blockIdx.y, s = blockIdx.x, c = threadIdx.x;
    const int* mrow = matrix + ((long)b * S_ + s) * W_;
    float base = P1s[((long)b * S_ + s) * 192 + c];
    float tv1 = t1[1 * 192 + c], tv2 = t1[2 * 192 + c], tv3 = t1[3 * 192 + c];
    float acc = 0.f;
    for (int w = 0; w < W_; ++w) {
        int m = mrow[w];
        if (m == 0) continue;
        float tv = (m == 1) ? tv1 : (m == 2 ? tv2 : tv3);
        float g  = sigm_(base + P1e[((long)b * W_ + w) * 192 + c] + tv);
        float val = (c < EH_) ? e_h[((long)b * W_ + w) * EH_ + c]
                              : ee[m * L_ + (c - EH_)];
        acc += val * g;
    }
    out[((long)b * S_ + s) * 192 + c] = acc;
}

// e_neigh_s[b,w,c] = sum_s mask * sigmoid(P2e[b,w,c]+P2s[b,s,c]+t2[m][c]) * val
__global__ __launch_bounds__(320)
void agg_eneigh_s(const int* __restrict__ matrix, const float* __restrict__ s_h,
                  const float* __restrict__ P2s, const float* __restrict__ P2e,
                  const float* __restrict__ t2, const float* __restrict__ ee,
                  float* __restrict__ out)
{
    int b = blockIdx.y, w = blockIdx.x, c = threadIdx.x;
    float base = P2e[((long)b * W_ + w) * 320 + c];
    float tv1 = t2[1 * 320 + c], tv2 = t2[2 * 320 + c], tv3 = t2[3 * 320 + c];
    float acc = 0.f;
    for (int s = 0; s < S_; ++s) {
        int m = matrix[((long)b * S_ + s) * W_ + w];
        if (m == 0) continue;
        float tv = (m == 1) ? tv1 : (m == 2 ? tv2 : tv3);
        float g  = sigm_(base + P2s[((long)b * S_ + s) * 320 + c] + tv);
        float val = (c < SH_) ? s_h[((long)b * S_ + s) * SH_ + c]
                              : ee[m * L_ + (c - SH_)];
        acc += val * g;
    }
    out[((long)b * W_ + w) * 320 + c] = acc;
}

// Xh_s f16 (1024 x 1472): [sent(512)|s_neigh_s(256)|s_neigh_e(192)|g(256)|s_h(256)]
__global__ void build_xh_s(const float* __restrict__ sent, const float* __restrict__ sns,
                           const float* __restrict__ sne, const float* __restrict__ g_h,
                           const float* __restrict__ s_h, _Float16* __restrict__ X)
{
    long i = (long)blockIdx.x * blockDim.x + threadIdx.x;
    if (i >= (long)B_ * S_ * 1472) return;
    int col = (int)(i % 1472); long r = i / 1472; int b = (int)(r / S_);
    float v;
    if      (col < 512)  v = sent[r * 512 + col];
    else if (col < 768)  v = sns[r * 256 + (col - 512)];
    else if (col < 960)  v = sne[r * 192 + (col - 768)];
    else if (col < 1216) v = g_h[b * 256 + (col - 960)];
    else                 v = s_h[r * 256 + (col - 1216)];
    X[i] = (_Float16)v;
}

// Xh_e f16 (1024 x 960): [word(256) | e_neigh_s(320) | g(256) | e_h(128)]
__global__ void build_xh_e(const float* __restrict__ word, const float* __restrict__ ens,
                           const float* __restrict__ g_h, const float* __restrict__ e_h,
                           _Float16* __restrict__ X)
{
    long i = (long)blockIdx.x * blockDim.x + threadIdx.x;
    if (i >= (long)B_ * W_ * 960) return;
    int col = (int)(i % 960); long r = i / 960; int b = (int)(r / W_);
    float v;
    if      (col < 256) v = word[r * 256 + col];
    else if (col < 576) v = ens[r * 320 + (col - 256)];
    else if (col < 832) v = g_h[b * 256 + (col - 576)];
    else                v = e_h[r * 128 + (col - 832)];
    X[i] = (_Float16)v;
}

// h <- z*h + (1-z)*tanh(C + r*U), masked. In-place on h; also emits f16 copy.
__global__ void gru_pointwise(const float* __restrict__ R, const float* __restrict__ Z,
                              const float* __restrict__ Cc, const float* __restrict__ U,
                              const int* __restrict__ mask, float* __restrict__ h,
                              _Float16* __restrict__ h16, int hd)
{
    int idx = blockIdx.x, c = threadIdx.x;
    long o = (long)idx * hd + c;
    float r = sigm_(R[o]);
    float z = sigm_(Z[o]);
    float u = tanhf(Cc[o] + r * U[o]);
    float nh = z * h[o] + (1.f - z) * u;
    if (mask[idx] == 0) nh = 0.f;
    h[o] = nh;
    h16[o] = (_Float16)nh;
}

__global__ __launch_bounds__(384)
void means_kernel(const float* __restrict__ s_h, const float* __restrict__ e_h,
                  const int* __restrict__ smask, const int* __restrict__ wmask,
                  float* __restrict__ sh_mean, float* __restrict__ eh_mean)
{
    int b = blockIdx.x, c = threadIdx.x;
    if (c < 256) {
        float acc = 0.f, cnt = 0.f;
        for (int s = 0; s < S_; ++s) {
            acc += s_h[((long)b * S_ + s) * 256 + c];
            cnt += (smask[b * S_ + s] != 0) ? 1.f : 0.f;
        }
        sh_mean[b * 256 + c] = acc / cnt;
    } else {
        int cc = c - 256;
        float acc = 0.f, cnt = 0.f;
        for (int w = 0; w < W_; ++w) {
            acc += e_h[((long)b * W_ + w) * 128 + cc];
            cnt += (wmask[b * W_ + w] != 0) ? 1.f : 0.f;
        }
        eh_mean[b * 128 + cc] = acc / (cnt + 1.f);
    }
}

// tiny global-state GRU, one block per batch, one thread per channel
__global__ __launch_bounds__(256)
void g_gru(const float* __restrict__ sh_mean, const float* __restrict__ eh_mean,
           const float* __restrict__ g_in,
           const float* __restrict__ rw, const float* __restrict__ rb,
           const float* __restrict__ zw, const float* __restrict__ zb,
           const float* __restrict__ cw, const float* __restrict__ cb,
           const float* __restrict__ uw, const float* __restrict__ ub,
           float* __restrict__ g_out)
{
    int b = blockIdx.x, c = threadIdx.x;
    __shared__ float xh[640];
    xh[c] = sh_mean[b * 256 + c];
    if (c < 128) xh[256 + c] = eh_mean[b * 128 + c];
    xh[384 + c] = g_in[b * 256 + c];
    __syncthreads();
    float r = rb[c], z = zb[c];
    for (int i = 0; i < 640; ++i) { float x = xh[i]; r += x * rw[i * 256 + c]; z += x * zw[i * 256 + c]; }
    r = sigm_(r); z = sigm_(z);
    float cv = cb[c];
    for (int i = 0; i < 384; ++i) cv += xh[i] * cw[i * 256 + c];
    float uv = ub[c];
    for (int i = 0; i < 256; ++i) uv += xh[384 + i] * uw[i * 256 + c];
    float u = tanhf(cv + r * uv);
    g_out[b * 256 + c] = z * xh[384 + c] + (1.f - z) * u;
}

extern "C" void kernel_launch(void* const* d_in, const int* in_sizes, int n_in,
                              void* d_out, int out_size, void* d_ws, size_t ws_size,
                              hipStream_t stream)
{
    (void)in_sizes; (void)n_in; (void)out_size; (void)ws_size;
    const float* sent  = (const float*)d_in[0];
    const float* word  = (const float*)d_in[1];
    const float* ee    = (const float*)d_in[2];
    const float* g1w   = (const float*)d_in[3];
    const float* g1b   = (const float*)d_in[4];
    const float* g2w   = (const float*)d_in[5];
    const float* g2b   = (const float*)d_in[6];
    const float* s_rw = (const float*)d_in[7];  const float* s_rb = (const float*)d_in[8];
    const float* s_zw = (const float*)d_in[9];  const float* s_zb = (const float*)d_in[10];
    const float* s_cw = (const float*)d_in[11]; const float* s_cb = (const float*)d_in[12];
    const float* s_uw = (const float*)d_in[13]; const float* s_ub = (const float*)d_in[14];
    const float* e_rw = (const float*)d_in[15]; const float* e_rb = (const float*)d_in[16];
    const float* e_zw = (const float*)d_in[17]; const float* e_zb = (const float*)d_in[18];
    const float* e_cw = (const float*)d_in[19]; const float* e_cb = (const float*)d_in[20];
    const float* e_uw = (const float*)d_in[21]; const float* e_ub = (const float*)d_in[22];
    const float* grw = (const float*)d_in[23]; const float* grb = (const float*)d_in[24];
    const float* gzw = (const float*)d_in[25]; const float* gzb = (const float*)d_in[26];
    const float* gcw = (const float*)d_in[27]; const float* gcb = (const float*)d_in[28];
    const float* guw = (const float*)d_in[29]; const float* gub = (const float*)d_in[30];
    const int* smask  = (const int*)d_in[31];
    const int* wmask  = (const int*)d_in[32];
    const int* matrix = (const int*)d_in[33];
    float* out = (float*)d_out;

    // ---- workspace carve-up: f32 region then f16 region ----
    float* w = (float*)d_ws;
    long off = 0;
    auto alloc = [&](long n) { float* p = w + off; off += (n + 63) & ~63L; return p; };
    float* t1    = alloc(4 * 192);
    float* t2    = alloc(4 * 320);
    float* s_h   = alloc((long)B_ * S_ * SH_);
    float* e_h   = alloc((long)B_ * W_ * EH_);
    float* gb0   = alloc((long)B_ * G_);
    float* gb1   = alloc((long)B_ * G_);
    float* P1s   = alloc(1024L * 192);
    float* P1e   = alloc(1024L * 192);
    float* P2s   = alloc(1024L * 320);
    float* P2e   = alloc(1024L * 320);
    float* sns   = alloc(1024L * 256);
    float* sne   = alloc(1024L * 192);
    float* ens   = alloc(1024L * 320);
    float* Rs = alloc(1024L * 256); float* Zs = alloc(1024L * 256);
    float* Cs = alloc(1024L * 256); float* Us = alloc(1024L * 256);
    float* Re = alloc(1024L * 128); float* Ze = alloc(1024L * 128);
    float* Ce = alloc(1024L * 128); float* Ue = alloc(1024L * 128);
    float* shm = alloc((long)B_ * 256);
    float* ehm = alloc((long)B_ * 128);

    _Float16* h16 = (_Float16*)(w + off);
    long hoff = 0;
    auto alloc16 = [&](long n) { _Float16* p = h16 + hoff; hoff += (n + 63) & ~63L; return p; };
    // B-fragment-major weights
    _Float16* F_p1s = alloc16(256L * 192);
    _Float16* F_p1e = alloc16(128L * 192);
    _Float16* F_p2s = alloc16(256L * 320);
    _Float16* F_p2e = alloc16(128L * 320);
    _Float16* F_srw = alloc16(1472L * 256);
    _Float16* F_szw = alloc16(1472L * 256);
    _Float16* F_scw = alloc16(1216L * 256);
    _Float16* F_suw = alloc16(256L * 256);
    _Float16* F_erw = alloc16(960L * 128);
    _Float16* F_ezw = alloc16(960L * 128);
    _Float16* F_ecw = alloc16(832L * 128);
    _Float16* F_euw = alloc16(128L * 128);
    _Float16* F_sh  = alloc16((long)B_ * S_ * SH_);   // s_h as B of s2s GEMM
    // f16 row-major A operands
    _Float16* s2s16  = alloc16((long)B_ * S_ * S_);
    _Float16* sh16   = alloc16((long)B_ * S_ * SH_);
    _Float16* eh16   = alloc16((long)B_ * W_ * EH_);
    _Float16* XhS16  = alloc16(1024L * 1472);
    _Float16* XhE16  = alloc16(1024L * 960);

    float* gbuf[2] = { gb0, gb1 };

    auto conv = [&](const float* W, _Float16* Wf, int K, int N, long sW, long sWf, int batch) {
        long n = (long)K * N;
        conv_wfrag<<<dim3((unsigned)((n + 255) / 256), 1, batch), 256, 0, stream>>>(
            W, Wf, K, N, sW, sWf);
    };
    auto gemm = [&](const _Float16* A, const _Float16* Bf, const float* bias, float* C,
                    int M, int N, int K, int lda, long sA, long sB, long sC, int batch) {
        gemm_wmma<<<dim3(N / 64, M / 64, batch), 128, 0, stream>>>(
            A, Bf, bias, C, M, N, K, lda, sA, sB, sC);
    };

    // ---- layer-invariant precompute ----
    build_s2s<<<dim3(S_, B_), 128, 0, stream>>>(matrix, smask, s2s16);
    build_tabs<<<4, 320, 0, stream>>>(ee, g1w, g1b, g2w, g2b, t1, t2);
    hipMemsetAsync(s_h, 0, (size_t)B_ * S_ * SH_ * 4, stream);
    hipMemsetAsync(e_h, 0, (size_t)B_ * W_ * EH_ * 4, stream);
    hipMemsetAsync(sh16, 0, (size_t)B_ * S_ * SH_ * 2, stream);
    hipMemsetAsync(eh16, 0, (size_t)B_ * W_ * EH_ * 2, stream);
    hipMemsetAsync(gb0, 0, (size_t)B_ * G_ * 4, stream);

    conv(g1w,              F_p1s, 256, 192, 0, 0, 1);
    conv(g1w + 256L * 192, F_p1e, 128, 192, 0, 0, 1);
    conv(g2w + 128L * 320, F_p2s, 256, 320, 0, 0, 1);
    conv(g2w,              F_p2e, 128, 320, 0, 0, 1);
    conv(s_rw, F_srw, 1472, 256, 0, 0, 1);
    conv(s_zw, F_szw, 1472, 256, 0, 0, 1);
    conv(s_cw, F_scw, 1216, 256, 0, 0, 1);
    conv(s_uw, F_suw, 256,  256, 0, 0, 1);
    conv(e_rw, F_erw, 960, 128, 0, 0, 1);
    conv(e_zw, F_ezw, 960, 128, 0, 0, 1);
    conv(e_cw, F_ecw, 832, 128, 0, 0, 1);
    conv(e_uw, F_euw, 128, 128, 0, 0, 1);

    for (int l = 0; l < 2; ++l) {
        float* gin  = gbuf[l & 1];
        float* gout = gbuf[(l + 1) & 1];

        // gate-logit projections (bias folded into t1/t2 tables)
        gemm(sh16, F_p1s, nullptr, P1s, 1024, 192, 256, 256, 0, 0, 0, 1);
        gemm(eh16, F_p1e, nullptr, P1e, 1024, 192, 128, 128, 0, 0, 0, 1);
        gemm(sh16, F_p2s, nullptr, P2s, 1024, 320, 256, 256, 0, 0, 0, 1);
        gemm(eh16, F_p2e, nullptr, P2e, 1024, 320, 128, 128, 0, 0, 0, 1);

        // s_neigh_s = s2s @ s_h : s_h to B-fragment layout, batched WMMA GEMM
        conv(s_h, F_sh, S_, SH_, (long)S_ * SH_, (long)S_ * SH_, B_);
        gemm(s2s16, F_sh, nullptr, sns, S_, SH_, S_, S_,
             (long)S_ * S_, (long)S_ * SH_, (long)S_ * SH_, B_);

        // decomposed gated edge aggregations
        agg_sneigh_e<<<dim3(S_, B_), 192, 0, stream>>>(matrix, e_h, P1s, P1e, t1, ee, sne);
        agg_eneigh_s<<<dim3(W_, B_), 320, 0, stream>>>(matrix, s_h, P2s, P2e, t2, ee, ens);

        // concatenated GRU inputs (written directly as f16 GEMM-A)
        build_xh_s<<<(int)(((long)B_ * S_ * 1472 + 255) / 256), 256, 0, stream>>>(
            sent, sns, sne, gin, s_h, XhS16);
        build_xh_e<<<(int)(((long)B_ * W_ * 960 + 255) / 256), 256, 0, stream>>>(
            word, ens, gin, e_h, XhE16);

        // sentence GRU GEMMs (xh=1472, x=1216, h=256)
        gemm(XhS16,        F_srw, s_rb, Rs, 1024, 256, 1472, 1472, 0, 0, 0, 1);
        gemm(XhS16,        F_szw, s_zb, Zs, 1024, 256, 1472, 1472, 0, 0, 0, 1);
        gemm(XhS16,        F_scw, s_cb, Cs, 1024, 256, 1216, 1472, 0, 0, 0, 1);
        gemm(XhS16 + 1216, F_suw, s_ub, Us, 1024, 256, 256,  1472, 0, 0, 0, 1);
        // entity GRU GEMMs (xh=960, x=832, h=128)
        gemm(XhE16,       F_erw, e_rb, Re, 1024, 128, 960, 960, 0, 0, 0, 1);
        gemm(XhE16,       F_ezw, e_zb, Ze, 1024, 128, 960, 960, 0, 0, 0, 1);
        gemm(XhE16,       F_ecw, e_cb, Ce, 1024, 128, 832, 960, 0, 0, 0, 1);
        gemm(XhE16 + 832, F_euw, e_ub, Ue, 1024, 128, 128, 960, 0, 0, 0, 1);

        gru_pointwise<<<1024, 256, 0, stream>>>(Rs, Zs, Cs, Us, smask, s_h, sh16, 256);
        gru_pointwise<<<1024, 128, 0, stream>>>(Re, Ze, Ce, Ue, wmask, e_h, eh16, 128);

        means_kernel<<<B_, 384, 0, stream>>>(s_h, e_h, smask, wmask, shm, ehm);
        g_gru<<<B_, 256, 0, stream>>>(shm, ehm, gin, grw, grb, gzw, gzb,
                                      gcw, gcb, guw, gub, gout);
    }

    // outputs: (s_h, e_h, g_h) flat; after 2 layers g lives in gbuf[0]
    hipMemcpyAsync(out,                       s_h,     (size_t)B_ * S_ * SH_ * 4,
                   hipMemcpyDeviceToDevice, stream);
    hipMemcpyAsync(out + (long)B_ * S_ * SH_, e_h,     (size_t)B_ * W_ * EH_ * 4,
                   hipMemcpyDeviceToDevice, stream);
    hipMemcpyAsync(out + (long)B_ * S_ * SH_ + (long)B_ * W_ * EH_, gbuf[0],
                   (size_t)B_ * G_ * 4, hipMemcpyDeviceToDevice, stream);
}
